// NMT_RNNG_81664508166623
// MI455X (gfx1250) — compile-verified, matrix operand order checked by reading, not run on hardware
//
#include <hip/hip_runtime.h>
#include <hip/hip_bf16.h>

// ---------------------------------------------------------------------------
// Types for WMMA
// ---------------------------------------------------------------------------
typedef __attribute__((ext_vector_type(16))) __bf16 v16bf;
typedef __attribute__((ext_vector_type(8)))  __bf16 v8bf;
typedef __attribute__((ext_vector_type(8)))  float  v8f;

__device__ inline __bf16 f2bf(float f) {
  unsigned u = __builtin_bit_cast(unsigned, f);
  u += 0x7fffu + ((u >> 16) & 1u);           // round-to-nearest-even
  unsigned short h = (unsigned short)(u >> 16);
  return __builtin_bit_cast(__bf16, h);
}

__device__ inline float sigf(float x) { return 1.0f / (1.0f + expf(-x)); }

__device__ inline float wave_sum(float v) {
#pragma unroll
  for (int o = 16; o > 0; o >>= 1) v += __shfl_xor(v, o, 32);
  return v;
}
__device__ inline float wave_max(float v) {
#pragma unroll
  for (int o = 16; o > 0; o >>= 1) v = fmaxf(v, __shfl_xor(v, o, 32));
  return v;
}

// ---------------------------------------------------------------------------
// Encoder: one block per direction, persistent over 256 timesteps.
// W matrices are [2048,512] row-major; x,h,c in LDS.
// ---------------------------------------------------------------------------
__global__ __launch_bounds__(1024) void enc_kernel(
    const int* __restrict__ src, const float* __restrict__ src_emb,
    const float* __restrict__ Wih_f, const float* __restrict__ Whh_f, const float* __restrict__ b_f,
    const float* __restrict__ Wih_b, const float* __restrict__ Whh_b, const float* __restrict__ b_b,
    const float* __restrict__ enc_hidden, float* __restrict__ enc_out) {
  const int dir = blockIdx.x;  // 0 = forward, 1 = backward
  const float* Wih = dir ? Wih_b : Wih_f;
  const float* Whh = dir ? Whh_b : Whh_f;
  const float* b   = dir ? b_b   : b_f;

  __shared__ float x[512], h[512], c[512], g[2048];
  const int tid = threadIdx.x;

  if (tid < 512) {
    h[tid] = enc_hidden[dir * 512 + tid];
    c[tid] = enc_hidden[1024 + dir * 512 + tid];
  }
  __syncthreads();

  for (int t = 0; t < 256; ++t) {
    const int idx = dir ? (255 - t) : t;
    if (tid < 512) x[tid] = src_emb[(long)src[idx] * 512 + tid];
    __syncthreads();
#pragma unroll
    for (int rr = 0; rr < 2; ++rr) {
      const int r = tid + rr * 1024;
      const float* wi = Wih + (long)r * 512;
      const float* wh = Whh + (long)r * 512;
      float acc = b[r];
      for (int j = 0; j < 512; ++j) acc = fmaf(wi[j], x[j], acc);
      for (int j = 0; j < 512; ++j) acc = fmaf(wh[j], h[j], acc);
      g[r] = acc;
    }
    __syncthreads();
    if (tid < 512) {
      const float ig = sigf(g[tid]);
      const float fg = sigf(g[512 + tid]);
      const float gg = tanhf(g[1024 + tid]);
      const float og = sigf(g[1536 + tid]);
      const float cn = fg * c[tid] + ig * gg;
      const float hn = og * tanhf(cn);
      c[tid] = cn; h[tid] = hn;
      enc_out[(long)idx * 1024 + dir * 512 + tid] = hn;  // fwd->cols 0..511, bwd->512..1023
    }
    __syncthreads();
  }
}

// ---------------------------------------------------------------------------
// Attention helper: all 1024 threads participate. dech[512] -> ctx[1024].
// ---------------------------------------------------------------------------
__device__ void attend_block(int tid, const float* __restrict__ enc_out,
                             const float* __restrict__ attn_W, const float* __restrict__ attn_b,
                             const float* dech, float* q, float* sc, float* red, float* ctx) {
  // q = attn_W @ dec_h + attn_b   (1024 x 512)
  {
    const float* w = attn_W + (long)tid * 512;
    float acc = attn_b[tid];
    for (int j = 0; j < 512; ++j) acc = fmaf(w[j], dech[j], acc);
    q[tid] = acc;
  }
  __syncthreads();
  // scores[t] = enc_out[t] . q   (256 dots of length 1024) — 8 per wave
  const int wave = tid >> 5, lane = tid & 31;
#pragma unroll
  for (int k = 0; k < 8; ++k) {
    const int t = wave * 8 + k;
    const float* row = enc_out + (long)t * 1024;
    float p = 0.f;
    for (int m = 0; m < 32; ++m) { const int j = lane + (m << 5); p = fmaf(row[j], q[j], p); }
    p = wave_sum(p);
    if (lane == 0) sc[t] = p;
  }
  __syncthreads();
  // softmax over 256 scores (wave 0)
  if (wave == 0) {
    float mx = -1e30f;
#pragma unroll
    for (int k = 0; k < 8; ++k) mx = fmaxf(mx, sc[lane + 32 * k]);
    mx = wave_max(mx);
    float s = 0.f;
#pragma unroll
    for (int k = 0; k < 8; ++k) { const float e = expf(sc[lane + 32 * k] - mx); sc[lane + 32 * k] = e; s += e; }
    s = wave_sum(s);
    if (lane == 0) red[0] = 1.0f / s;
  }
  __syncthreads();
  const float inv = red[0];
  // ctx[j] = sum_t alpha[t] * enc_out[t][j]
  {
    float acc = 0.f;
    for (int t = 0; t < 256; ++t) acc = fmaf(sc[t], enc_out[(long)t * 1024 + tid], acc);
    ctx[tid] = acc * inv;
  }
  __syncthreads();
}

// ---------------------------------------------------------------------------
// Decoder: single persistent block runs init + 255 scan steps.
// ---------------------------------------------------------------------------
__global__ __launch_bounds__(1024) void dec_kernel(
    const int* __restrict__ tgt, const int* __restrict__ actions,
    const float* __restrict__ tgt_emb, const float* __restrict__ act_emb,
    const float* __restrict__ act_Wih, const float* __restrict__ act_Whh, const float* __restrict__ act_b,
    const float* __restrict__ dec_Wih, const float* __restrict__ dec_Whh, const float* __restrict__ dec_b,
    const float* __restrict__ out_Wih, const float* __restrict__ out_b,
    const float* __restrict__ decInit_W, const float* __restrict__ decInit_b,
    const float* __restrict__ attn_W, const float* __restrict__ attn_b,
    const float* __restrict__ stilde_W, const float* __restrict__ stilde_b,
    const float* __restrict__ ut_W, const float* __restrict__ ut_b,
    const float* __restrict__ enc_out,
    float* __restrict__ s_tildes, float* __restrict__ uts) {
  __shared__ float sx[512], el[1024], dech[512], decc[512];
  __shared__ float acth[128], actc[128], outh[512], st[512];
  __shared__ float g[2048], q[1024], ctx[1024], sc[256], red[2];
  const int tid = threadIdx.x;

  if (tid < 512) { dech[tid] = 0.f; decc[tid] = 0.f; st[tid] = 0.f; outh[tid] = 0.f; }
  if (tid < 128) { acth[tid] = 0.f; actc[tid] = 0.f; }
  // enc_last = concat(hs_f[255], hs_b[first]) = [enc_out[255][0:512], enc_out[0][512:1024]]
  el[tid] = (tid < 512) ? enc_out[(long)255 * 1024 + tid] : enc_out[tid];
  if (tid < 128) sx[tid] = act_emb[(long)actions[0] * 128 + tid];
  __syncthreads();

  // --- act StackLSTM step 0 (h=c=0) ---
  if (tid < 512) {
    const float* w = act_Wih + (long)tid * 128;
    float acc = act_b[tid];
    for (int j = 0; j < 128; ++j) acc = fmaf(w[j], sx[j], acc);
    g[tid] = acc;
  }
  __syncthreads();
  if (tid < 128) {
    const float ig = sigf(g[tid]);
    const float gg = tanhf(g[256 + tid]), og = sigf(g[384 + tid]);
    const float cn = ig * gg;            // c was 0
    actc[tid] = cn; acth[tid] = og * tanhf(cn);
  }
  if (tid < 512) sx[tid] = tgt_emb[(long)tgt[0] * 512 + tid];
  __syncthreads();

  // --- outBuf StackLSTM step 0 (h=c=0); out_h constant afterwards ---
#pragma unroll
  for (int rr = 0; rr < 2; ++rr) {
    const int r = tid + rr * 1024;
    const float* w = out_Wih + (long)r * 512;
    float acc = out_b[r];
    for (int j = 0; j < 512; ++j) acc = fmaf(w[j], sx[j], acc);
    g[r] = acc;
  }
  __syncthreads();
  if (tid < 512) {
    const float ig = sigf(g[tid]), og = sigf(g[1536 + tid]);
    const float gg = tanhf(g[1024 + tid]);
    const float cn = ig * gg;
    outh[tid] = og * tanhf(cn);
  }
  __syncthreads();

  // --- decoder init: dec_h = tanh(decInit_W @ enc_last + b), dec_c = 0 ---
  if (tid < 512) {
    const float* w = decInit_W + (long)tid * 1024;
    float acc = decInit_b[tid];
    for (int j = 0; j < 1024; ++j) acc = fmaf(w[j], el[j], acc);
    dech[tid] = tanhf(acc);
  }
  __syncthreads();

  attend_block(tid, enc_out, attn_W, attn_b, dech, q, sc, red, ctx);

  // s_tilde = tanh(stilde_W @ [dec_h; ctx] + b)
  if (tid < 512) {
    const float* w = stilde_W + (long)tid * 1536;
    float acc = stilde_b[tid];
    for (int j = 0; j < 512;  ++j) acc = fmaf(w[j],        dech[j], acc);
    for (int j = 0; j < 1024; ++j) acc = fmaf(w[512 + j],  ctx[j],  acc);
    st[tid] = tanhf(acc);
  }
  // ut0 = tanh(ut_W @ [dec_h; out_h; act_h] + b)
  if (tid < 512) {
    const float* w = ut_W + (long)tid * 1152;
    float acc = ut_b[tid];
    for (int j = 0; j < 512; ++j) acc = fmaf(w[j],         dech[j], acc);
    for (int j = 0; j < 512; ++j) acc = fmaf(w[512 + j],   outh[j], acc);
    for (int j = 0; j < 128; ++j) acc = fmaf(w[1024 + j],  acth[j], acc);
    uts[tid] = tanhf(acc);
  }
  __syncthreads();

  // --- scan: steps i = 1..255, a_in = act_emb[actions[i-1]] ---
  for (int i = 1; i < 256; ++i) {
    if (tid < 128) sx[tid] = act_emb[(long)actions[i - 1] * 128 + tid];
    __syncthreads();
    // act cell
    if (tid < 512) {
      const float* wi = act_Wih + (long)tid * 128;
      const float* wh = act_Whh + (long)tid * 128;
      float acc = act_b[tid];
      for (int j = 0; j < 128; ++j) acc = fmaf(wi[j], sx[j], acc);
      for (int j = 0; j < 128; ++j) acc = fmaf(wh[j], acth[j], acc);
      g[tid] = acc;
    }
    __syncthreads();
    if (tid < 128) {
      const float ig = sigf(g[tid]), fg = sigf(g[128 + tid]);
      const float gg = tanhf(g[256 + tid]), og = sigf(g[384 + tid]);
      const float cn = fg * actc[tid] + ig * gg;
      actc[tid] = cn; acth[tid] = og * tanhf(cn);
    }
    __syncthreads();
    // dec cell: input = previous s_tilde
#pragma unroll
    for (int rr = 0; rr < 2; ++rr) {
      const int r = tid + rr * 1024;
      const float* wi = dec_Wih + (long)r * 512;
      const float* wh = dec_Whh + (long)r * 512;
      float acc = dec_b[r];
      for (int j = 0; j < 512; ++j) acc = fmaf(wi[j], st[j], acc);
      for (int j = 0; j < 512; ++j) acc = fmaf(wh[j], dech[j], acc);
      g[r] = acc;
    }
    __syncthreads();
    if (tid < 512) {
      const float ig = sigf(g[tid]), fg = sigf(g[512 + tid]);
      const float gg = tanhf(g[1024 + tid]), og = sigf(g[1536 + tid]);
      const float cn = fg * decc[tid] + ig * gg;
      decc[tid] = cn; dech[tid] = og * tanhf(cn);
    }
    __syncthreads();

    attend_block(tid, enc_out, attn_W, attn_b, dech, q, sc, red, ctx);

    if (tid < 512) {
      const float* w = stilde_W + (long)tid * 1536;
      float acc = stilde_b[tid];
      for (int j = 0; j < 512;  ++j) acc = fmaf(w[j],       dech[j], acc);
      for (int j = 0; j < 1024; ++j) acc = fmaf(w[512 + j], ctx[j],  acc);
      const float v = tanhf(acc);
      st[tid] = v;
      s_tildes[(long)(i - 1) * 512 + tid] = v;
    }
    if (tid < 512) {
      const float* w = ut_W + (long)tid * 1152;
      float acc = ut_b[tid];
      for (int j = 0; j < 512; ++j) acc = fmaf(w[j],        dech[j], acc);
      for (int j = 0; j < 512; ++j) acc = fmaf(w[512 + j],  outh[j], acc);
      for (int j = 0; j < 128; ++j) acc = fmaf(w[1024 + j], acth[j], acc);
      uts[(long)i * 512 + tid] = tanhf(acc);
    }
    __syncthreads();
  }
}

// ---------------------------------------------------------------------------
// WMMA GEMM: C[M<=256, N] = A[M,512] @ B[N,512]^T + bias[N].
// Each wave owns one 16-col N-tile and ALL 16 M-tiles (16 x v8f accumulators),
// so each converted B fragment feeds 16 v_wmma ops and B is read exactly once.
// The A K-slice (256 rows x 32 K) is cooperatively converted to bf16 in LDS
// (padded row stride 40 halves -> conflict-free) and consumed via ds_load_b128.
// ---------------------------------------------------------------------------
#define AS_STRIDE 40  // halves per row (32 data + 8 pad), 80B keeps 16B align

__device__ inline v16bf load_frag_bf16_g(const float* __restrict__ p) {
  v16bf r;
#pragma unroll
  for (int i = 0; i < 8; ++i) r[i] = f2bf(p[i]);
#pragma unroll
  for (int i = 0; i < 8; ++i) r[8 + i] = f2bf(p[16 + i]);
  return r;
}

__global__ __launch_bounds__(128) void gemm_bias_wmma(
    const float* __restrict__ A, const float* __restrict__ B,
    const float* __restrict__ bias, float* __restrict__ C,
    int Mvalid, int N) {
  __shared__ __bf16 As[256 * AS_STRIDE];   // 20 KB

  const int tid   = threadIdx.x;
  const int lane  = tid & 31;
  const int wave  = tid >> 5;
  const int tileN = blockIdx.x * 4 + wave;
  const int n     = tileN * 16 + (lane & 15);  // B column / output column
  const bool hiK  = (lane >= 16);
  const int  kofs = hiK ? 8 : 0;               // fragment K phase per ISA layout

  const bool bValid = (n < N);
  const float bv = bValid ? bias[n] : 0.f;

  v8f acc[16];
#pragma unroll
  for (int mt = 0; mt < 16; ++mt)
#pragma unroll
    for (int v = 0; v < 8; ++v) acc[mt][v] = bv;  // bias folded into C

  v16bf zf;
#pragma unroll
  for (int i = 0; i < 16; ++i) zf[i] = __builtin_bit_cast(__bf16, (unsigned short)0);

  const float* pb = B + (long)n * 512;
  const int rbase = tid >> 5;     // 0..3
  const int kcol  = tid & 31;     // 0..31

  for (int kk = 0; kk < 512; kk += 32) {
    // ---- stage A slice [256 x 32] as bf16 into LDS (coalesced f32 reads) ----
#pragma unroll
    for (int it = 0; it < 64; ++it) {
      const int r = rbase + it * 4;             // 0..255
      const float av = (r < Mvalid) ? A[(long)r * 512 + kk + kcol] : 0.f;
      As[r * AS_STRIDE + kcol] = f2bf(av);
    }
    __syncthreads();

    // ---- B fragment: load f32, convert once, reuse for 16 M-tiles ----
    __builtin_prefetch(pb + kk + 32, 0, 1);     // next K chunk
    const v16bf bf = bValid ? load_frag_bf16_g(pb + kk + kofs) : zf;

#pragma unroll
    for (int mt = 0; mt < 16; ++mt) {
      const __bf16* arow = As + (mt * 16 + (lane & 15)) * AS_STRIDE + kofs;
      const v8bf lo = *(const v8bf*)(arow);
      const v8bf hi = *(const v8bf*)(arow + 16);
      const v16bf af = __builtin_shufflevector(lo, hi,
          0, 1, 2, 3, 4, 5, 6, 7, 8, 9, 10, 11, 12, 13, 14, 15);
      acc[mt] = __builtin_amdgcn_wmma_f32_16x16x32_bf16(
          /*neg_a=*/false, af, /*neg_b=*/false, bf,
          /*c_mod=*/(short)0, acc[mt], /*reuse_a=*/false, /*reuse_b=*/false);
    }
    __syncthreads();
  }

  if (bValid) {
#pragma unroll
    for (int mt = 0; mt < 16; ++mt) {
      const int mBase = mt * 16 + (hiK ? 8 : 0);  // D layout: lanes>=16 hold M=8..15
#pragma unroll
      for (int v = 0; v < 8; ++v) {
        const int m = mBase + v;
        if (m < Mvalid) C[(long)m * N + n] = acc[mt][v];
      }
    }
  }
}

// ---------------------------------------------------------------------------
// Launch
// ---------------------------------------------------------------------------
extern "C" void kernel_launch(void* const* d_in, const int* in_sizes, int n_in,
                              void* d_out, int out_size, void* d_ws, size_t ws_size,
                              hipStream_t stream) {
  (void)in_sizes; (void)n_in; (void)out_size; (void)ws_size;
  const int*   src        = (const int*)  d_in[0];
  const int*   tgt        = (const int*)  d_in[1];
  const int*   actions    = (const int*)  d_in[2];
  /* d_in[3] = src_length (unused, fixed 256) */
  const float* enc_hidden = (const float*)d_in[4];
  const float* src_emb    = (const float*)d_in[5];
  const float* tgt_emb    = (const float*)d_in[6];
  const float* act_emb    = (const float*)d_in[7];
  const float* enc_Wih_f  = (const float*)d_in[8];
  const float* enc_Whh_f  = (const float*)d_in[9];
  const float* enc_b_f    = (const float*)d_in[10];
  const float* enc_Wih_b  = (const float*)d_in[11];
  const float* enc_Whh_b  = (const float*)d_in[12];
  const float* enc_b_b    = (const float*)d_in[13];
  const float* dec_Wih    = (const float*)d_in[14];
  const float* dec_Whh    = (const float*)d_in[15];
  const float* dec_b      = (const float*)d_in[16];
  const float* act_Wih    = (const float*)d_in[17];
  const float* act_Whh    = (const float*)d_in[18];
  const float* act_b      = (const float*)d_in[19];
  const float* out_Wih    = (const float*)d_in[20];
  /* d_in[21] = out_Whh (multiplied by zero state; unused) */
  const float* out_b      = (const float*)d_in[22];
  const float* decInit_W  = (const float*)d_in[23];
  const float* decInit_b  = (const float*)d_in[24];
  const float* attn_W     = (const float*)d_in[25];
  const float* attn_b     = (const float*)d_in[26];
  const float* stilde_W   = (const float*)d_in[27];
  const float* stilde_b   = (const float*)d_in[28];
  const float* ut_W       = (const float*)d_in[29];
  const float* ut_b       = (const float*)d_in[30];
  const float* actPred_W  = (const float*)d_in[31];
  const float* actPred_b  = (const float*)d_in[32];
  const float* wordPred_W = (const float*)d_in[33];
  const float* wordPred_b = (const float*)d_in[34];

  char* ws = (char*)d_ws;
  float* enc_out  = (float*)ws;                                   // [256,1024]
  float* s_tildes = (float*)(ws + (size_t)256 * 1024 * 4);        // [255,512]
  float* uts      = (float*)(ws + (size_t)256 * 1024 * 4
                                + (size_t)255 * 512 * 4);         // [256,512]

  float* out          = (float*)d_out;
  float* act_logits   = out;               // [256,128]
  float* word_logits  = out + 256 * 128;   // [255,32000]

  enc_kernel<<<2, 1024, 0, stream>>>(src, src_emb,
                                     enc_Wih_f, enc_Whh_f, enc_b_f,
                                     enc_Wih_b, enc_Whh_b, enc_b_b,
                                     enc_hidden, enc_out);

  dec_kernel<<<1, 1024, 0, stream>>>(tgt, actions, tgt_emb, act_emb,
                                     act_Wih, act_Whh, act_b,
                                     dec_Wih, dec_Whh, dec_b,
                                     out_Wih, out_b,
                                     decInit_W, decInit_b,
                                     attn_W, attn_b,
                                     stilde_W, stilde_b,
                                     ut_W, ut_b,
                                     enc_out, s_tildes, uts);

  // action_logits = uts @ actPred_W^T + b  : M=256, N=128 (2 blocks x 4 N-tiles)
  gemm_bias_wmma<<<dim3(2), 128, 0, stream>>>(uts, actPred_W, actPred_b,
                                              act_logits, 256, 128);
  // word_logits = s_tildes @ wordPred_W^T + b : M=255, N=32000 (500 blocks)
  gemm_bias_wmma<<<dim3(500), 128, 0, stream>>>(s_tildes, wordPred_W, wordPred_b,
                                                word_logits, 255, 32000);
}
